// MDTNN_60241211293745
// MI455X (gfx1250) — compile-verified, matrix operand order checked by reading, not run on hardware
//
#include <hip/hip_runtime.h>
#include <hip/hip_bf16.h>

typedef __attribute__((ext_vector_type(16))) _Float16 v16h;
typedef __attribute__((ext_vector_type(8)))  _Float16 v8h;
typedef __attribute__((ext_vector_type(8)))  float    v8f;
typedef __attribute__((ext_vector_type(8)))  unsigned short v8us;

#define NBASIS   30
#define NHIDDEN  15
#define A_MAX    30
#define NPAIR    900
#define NPAIR_PAD 912   /* 57*16 */
#define NGAUSS   56
#define T_STEPS  3
#define NBATCH   2048

/* LDS layout (bytes) */
#define OFF_DHAT   0                                   /* [912][32] f16, A-operand layout */
#define OFF_DFB_W  (OFF_DHAT + NPAIR_PAD*32*2)         /* [32 cols][64 K] f16, df_w col-major (+df_b row 56) */
#define OFF_FCB_W  (OFF_DFB_W + 32*64*2)               /* [32 cols][32 K] f16, fc_w col-major */
#define OFF_C      (OFF_FCB_W + 32*32*2)               /* [30][32] f32 (ping) */
#define OFF_CACC   (OFF_C + 30*32*4)                   /* [30][32] f32 (pong) */
#define OFF_CFWT   (OFF_CACC + 30*32*4)                /* [30 k][32 h] f32, cf_w transposed */
#define OFF_XROW   (OFF_CFWT + 30*32*4)                /* [30][32] f16 */
#define OFF_CFB    (OFF_XROW + 30*32*2)                /* [32] f32 */
#define OFF_IOFF   (OFF_CFB + 32*4)                    /* [912] u16: min(p/30,29)*64  */
#define OFF_JOFF   (OFF_IOFF + NPAIR_PAD*2)            /* [912] u16: (p%30)*128       */
#define OFF_PMASK  (OFF_JOFF + NPAIR_PAD*2)            /* [912] f16: validity mask    */
#define OFF_RED    (OFF_PMASK + NPAIR_PAD*2)           /* [32] f32 */
#define LDS_BYTES  (OFF_RED + 32*4)                    /* = 83680 */

__device__ __forceinline__ float fast_tanh(float x) {
  /* tanh(x) = 1 - 2/(exp(2x)+1); exact 0 at x=0, saturates to +-1 correctly */
  float ex = __expf(x + x);
  return 1.0f - __fdividef(2.0f, ex + 1.0f);
}

__global__ __launch_bounds__(256) void MDTNN_fused_wmma(
    const int* __restrict__ Z, const float* __restrict__ D,
    const int* __restrict__ sizes, const float* __restrict__ C_embed,
    const float* __restrict__ df_w, const float* __restrict__ df_b,
    const float* __restrict__ cf_w, const float* __restrict__ cf_b,
    const float* __restrict__ fc_w, const float* __restrict__ mlp_w1,
    const float* __restrict__ mlp_b1, const float* __restrict__ mlp_w2,
    const float* __restrict__ mlp_b2, float* __restrict__ out)
{
  extern __shared__ char smem[];
  _Float16* s_dhat  = (_Float16*)(smem + OFF_DHAT);
  _Float16* s_dfB   = (_Float16*)(smem + OFF_DFB_W);
  _Float16* s_fcB   = (_Float16*)(smem + OFF_FCB_W);
  float*    s_C0    = (float*)   (smem + OFF_C);
  float*    s_C1    = (float*)   (smem + OFF_CACC);
  float*    s_cfwT  = (float*)   (smem + OFF_CFWT);
  _Float16* s_xrow  = (_Float16*)(smem + OFF_XROW);
  float*    s_cfb   = (float*)   (smem + OFF_CFB);
  unsigned short* s_ioff = (unsigned short*)(smem + OFF_IOFF);
  unsigned short* s_joff = (unsigned short*)(smem + OFF_JOFF);
  _Float16* s_pmask = (_Float16*)(smem + OFF_PMASK);
  float*    s_red   = (float*)   (smem + OFF_RED);

  const int tid    = threadIdx.x;
  const int lane   = tid & 31;          /* wave32 */
  const int wave   = tid >> 5;          /* 8 waves */
  const int lcol   = lane & 15;
  const int half16 = (lane < 16) ? 0 : 16;
  const int hi8    = (lane < 16) ? 0 : 8;
  const int b      = blockIdx.x;
  const int size   = sizes[b];

  /* ---------------- phase 0: stage weights / embeddings / tables into LDS ---- */
  for (int idx = tid; idx < 30*32; idx += 256) {
    int j = idx >> 5, k = idx & 31;
    float v = 0.f;
    if (k < NBASIS) v = C_embed[Z[b*A_MAX + j]*NBASIS + k];
    s_C0[idx] = v;
  }
  for (int idx = tid; idx < 32*64; idx += 256) {   /* df_w col-major; row 56 = df_b */
    int n = idx >> 6, g = idx & 63;
    float v = 0.f;
    if (n < NBASIS) {
      if (g < NGAUSS)       v = df_w[g*NBASIS + n];
      else if (g == NGAUSS) v = df_b[n];
    }
    s_dfB[idx] = (_Float16)v;
  }
  for (int idx = tid; idx < 32*32; idx += 256) {   /* fc_w col-major, zero-padded */
    int n = idx >> 5, h = idx & 31;
    float v = (n < NBASIS && h < NBASIS) ? fc_w[h*NBASIS + n] : 0.f;
    s_fcB[idx] = (_Float16)v;
  }
  for (int idx = tid; idx < 30*32; idx += 256) {   /* cf_w TRANSPOSED: [k][h] */
    int k = idx >> 5, h = idx & 31;
    s_cfwT[idx] = (h < NBASIS) ? cf_w[h*NBASIS + k] : 0.f;
  }
  for (int idx = tid; idx < NPAIR_PAD; idx += 256) {  /* per-pair tables (once) */
    int i = idx / A_MAX;
    int j = idx - i*A_MAX;
    int ic = (i > 29) ? 29 : i;
    s_ioff[idx] = (unsigned short)(ic * 64);          /* byte offset into s_xrow */
    s_joff[idx] = (unsigned short)(j * 128);          /* byte offset into C rows */
    float m = (idx < NPAIR && i < size && j < size) ? 1.f : 0.f;
    s_pmask[idx] = (_Float16)m;
  }
  if (tid < 32) s_cfb[tid] = (tid < NBASIS) ? cf_b[tid] : 0.f;
  __syncthreads();

  /* hoisted B operands for phase 1 (invariant across tiles) */
  v16h dfb0[2], dfb1[2];
  #pragma unroll
  for (int nt = 0; nt < 2; ++nt) {
    int col = nt*16 + lcol;
    dfb0[nt] = *(const v16h*)(s_dfB + col*64 + half16);
    dfb1[nt] = *(const v16h*)(s_dfB + col*64 + 32 + half16);
  }

  /* ---------------- phase 1: d_hat = D[b] @ df_w (+bias row)  (WMMA K=64) ----
     D read ONCE from HBM; result masked via pmask and kept in LDS as f16.
     Masked / pad rows stored as EXACT zeros => T-loop needs no masks and the
     pad atomics below add exactly tanh(0)=0 (fc has no bias). ----------------- */
  for (int t = wave; t < 57; t += 8) {
    int p = 16*t + lcol;                       /* pair index this lane supplies */
    bool pv = (p < NPAIR);
    const float* Drow = D + ((size_t)b*NPAIR + (size_t)(pv ? p : 0))*NGAUSS;
    v16h a0, a1;
    if (pv) {
      #pragma unroll
      for (int e = 0; e < 16; ++e) a0[e] = (_Float16)Drow[half16 + e];
      #pragma unroll
      for (int e = 0; e < 16; ++e) {
        int kk = 32 + half16 + e;
        float v = (kk < NGAUSS) ? Drow[kk] : ((kk == NGAUSS) ? 1.0f : 0.f);
        a1[e] = (_Float16)v;                   /* k=56 carries the bias 1-column */
      }
    } else {
      #pragma unroll
      for (int e = 0; e < 16; ++e) { a0[e] = (_Float16)0.f; a1[e] = (_Float16)0.f; }
    }
    const v8h mv = *(const v8h*)(s_pmask + 16*t + hi8);   /* 8 row masks: 1 load */
    #pragma unroll
    for (int nt = 0; nt < 2; ++nt) {
      v8f c = {};
      c = __builtin_amdgcn_wmma_f32_16x16x32_f16(false, a0, false, dfb0[nt], (short)0, c, false, false);
      c = __builtin_amdgcn_wmma_f32_16x16x32_f16(false, a1, false, dfb1[nt], (short)0, c, false, false);
      _Float16* sp = s_dhat + (16*t + hi8)*32 + nt*16 + lcol;
      #pragma unroll
      for (int r = 0; r < 8; ++r) {
        _Float16 hv = (_Float16)c[r];
        *sp = hv * mv[r];                      /* cvt + pk-mul + b16 store only  */
        sp += 32;
      }
    }
  }
  __syncthreads();

  /* hoisted fc_w B operands for phase 2 (invariant across tiles and steps) */
  v16h fcb[2];
  #pragma unroll
  for (int nt = 0; nt < 2; ++nt)
    fcb[nt] = *(const v16h*)(s_fcB + (nt*16 + lcol)*32 + half16);

  /* ---------------- phase 2: T message-passing steps entirely in LDS --------- */
  float* cur = s_C0;
  float* nxt = s_C1;
  for (int step = 0; step < T_STEPS; ++step) {
    /* Xrow = C @ cf_w + cf_b (tiny, contiguous b128 reads); seed nxt with C */
    for (int idx = tid; idx < 30*32; idx += 256) {
      int i = idx >> 5, k = idx & 31;
      float acc = 0.f;
      if (k < NBASIS) {
        acc = s_cfb[k];
        const float* crow = cur    + i*32;
        const float* wrow = s_cfwT + k*32;
        #pragma unroll
        for (int h = 0; h < NBASIS; ++h) acc += crow[h] * wrow[h];
      }
      s_xrow[idx] = (_Float16)acc;
      nxt[idx] = cur[idx];
    }
    __syncthreads();

    /* Y = tanh((Xrow_i * d_hat_ij) @ fc_w); C[j] += sum_i Y  (WMMA + ds_add) */
    for (int t = wave; t < 57; t += 8) {
      int p = 16*t + lcol;
      v16h d = *(const v16h*)(s_dhat + p*32 + half16);
      int ioffb = (int)s_ioff[p];                        /* table: no div/clamp  */
      v16h x = *(const v16h*)((const char*)s_xrow + ioffb + half16*2);
      v16h a = d * x;
      const v8us jv = *(const v8us*)(s_joff + 16*t + hi8);  /* 8 row offs: 1 load */
      #pragma unroll
      for (int nt = 0; nt < 2; ++nt) {
        v8f c = {};
        c = __builtin_amdgcn_wmma_f32_16x16x32_f16(false, a, false, fcb[nt], (short)0, c, false, false);
        const int k4 = (nt*16 + lcol) * 4;
        #pragma unroll
        for (int r = 0; r < 8; ++r) {
          /* unconditional: masked/pad rows contribute exactly tanh(0)=0 */
          atomicAdd((float*)((char*)nxt + (int)jv[r] + k4), fast_tanh(c[r]));
        }
      }
    }
    __syncthreads();
    float* tmp = cur; cur = nxt; nxt = tmp;              /* ping-pong, no copy */
  }

  /* ---------------- phase 3: readout MLP + masked reduction ------------------ */
  if (tid < A_MAX) {
    float e = mlp_b2[0];
    #pragma unroll
    for (int hh = 0; hh < NHIDDEN; ++hh) {
      float acc = mlp_b1[hh];
      #pragma unroll 6
      for (int k = 0; k < NBASIS; ++k) acc += cur[tid*32 + k] * mlp_w1[k*NHIDDEN + hh];
      e += fast_tanh(acc) * mlp_w2[hh];
    }
    s_red[tid] = (tid < size) ? e : 0.f;
  }
  __syncthreads();
  if (tid == 0) {
    float s = 0.f;
    #pragma unroll
    for (int j = 0; j < A_MAX; ++j) s += s_red[j];
    out[b] = s;
  }
}

extern "C" void kernel_launch(void* const* d_in, const int* in_sizes, int n_in,
                              void* d_out, int out_size, void* d_ws, size_t ws_size,
                              hipStream_t stream) {
  const int*   Z      = (const int*)  d_in[0];
  const float* D      = (const float*)d_in[1];
  const int*   sizes  = (const int*)  d_in[2];
  const float* C_emb  = (const float*)d_in[3];
  const float* df_w   = (const float*)d_in[4];
  const float* df_b   = (const float*)d_in[5];
  const float* cf_w   = (const float*)d_in[6];
  const float* cf_b   = (const float*)d_in[7];
  const float* fc_w   = (const float*)d_in[8];
  const float* mlp_w1 = (const float*)d_in[9];
  const float* mlp_b1 = (const float*)d_in[10];
  const float* mlp_w2 = (const float*)d_in[11];
  const float* mlp_b2 = (const float*)d_in[12];
  float* out = (float*)d_out;

  (void)in_sizes; (void)n_in; (void)out_size; (void)d_ws; (void)ws_size;

  hipFuncSetAttribute((const void*)MDTNN_fused_wmma,
                      hipFuncAttributeMaxDynamicSharedMemorySize, LDS_BYTES);
  MDTNN_fused_wmma<<<NBATCH, 256, LDS_BYTES, stream>>>(
      Z, D, sizes, C_emb, df_w, df_b, cf_w, cf_b, fc_w,
      mlp_w1, mlp_b1, mlp_w2, mlp_b2, out);
}
